// VQModule_20255065767997
// MI455X (gfx1250) — matrix operand order repaired and checked by reading, not compiled, not run on hardware
//
#include <hip/hip_runtime.h>
#include <hip/hip_bf16.h>
#include <cstdint>

// ---- problem constants ----
#define BB     16
#define CC     256
#define HH     32
#define WW     32
#define KCODES 8192
#define NTOK   (BB * HH * WW)           // 16384 tokens
#define QE_ELEMS (BB * CC * HH * WW)    // 4194304
#define LOSS_OFF QE_ELEMS
#define IDX_OFF  (QE_ELEMS + 1)

// K-dimension split for occupancy: 8 segments of 1024 codes.
#define KSPLIT 8
#define KSEG   (KCODES / KSPLIT)        // 1024
#define TCODES 16                       // codes staged per iteration (1 N-tile)
#define NITER  (KSEG / TCODES)          // 64

// Tokens per wave (two register-resident 16-row A tile sets) -> each B
// fragment read from LDS feeds 2 WMMAs, halving LDS read traffic per MAC.
#define MWAVE  32
#define BLK_TOK (8 * MWAVE)             // 256 tokens per 256-thread block

// LDS row stride for a code row (pad 8 halves = 16B so the 16 lanes of a
// half-group hit distinct bank quads: 264*2/4 = 132 == 4 mod 64).
#define BROW        264
#define TILE_HALVES (TCODES * BROW)     // 4224
#define TILE_BYTES  (TILE_HALVES * 2)   // 8448

typedef __attribute__((ext_vector_type(16))) _Float16 v16h;
typedef __attribute__((ext_vector_type(8)))  _Float16 v8h;
typedef __attribute__((ext_vector_type(8)))  float    v8f;

union ABfrag { v16h v; v8h h[2]; };

// ---------------------------------------------------------------------------
// Kernel 1: x [B,C,H,W] f32 -> xT [N][C] f16 holding (-2*x), token-major.
// The -2 factor folds the "-2 x.e" distance term into the WMMA itself
// (x2 scaling is exact in f16). Reads coalesced.
// ---------------------------------------------------------------------------
__global__ __launch_bounds__(256) void prep_x(const float* __restrict__ x,
                                              _Float16* __restrict__ xT) {
    int i = blockIdx.x * 256 + threadIdx.x;         // flat over B*C*H*W
    int w = i & 31;
    int h = (i >> 5) & 31;
    int c = (i >> 10) & 255;
    int b = i >> 18;
    int n = (b << 10) + (h << 5) + w;               // token id
    xT[(size_t)n * CC + c] = (_Float16)(-2.0f * x[i]);
}

// ---------------------------------------------------------------------------
// Kernel 2: codebook f32 -> f16 copy + per-code squared norms.
// One wave32 per code; 8 codes per 256-thread block. Also zero the loss slot.
// ---------------------------------------------------------------------------
__global__ __launch_bounds__(256) void prep_cb(const float* __restrict__ cb,
                                               _Float16* __restrict__ cbh,
                                               float* __restrict__ cnorm,
                                               float* __restrict__ lossSlot) {
    const int lane = threadIdx.x & 31;
    const int wid  = threadIdx.x >> 5;
    const int k    = blockIdx.x * 8 + wid;
    const float* row = cb + (size_t)k * CC;
    float s = 0.0f;
#pragma unroll
    for (int j = lane; j < CC; j += 32) {
        float v = row[j];
        cbh[(size_t)k * CC + j] = (_Float16)v;
        s += v * v;
    }
#pragma unroll
    for (int m = 16; m >= 1; m >>= 1) s += __shfl_xor(s, m, 32);
    if (lane == 0) cnorm[k] = s;
    if (blockIdx.x == 0 && threadIdx.x == 0) *lossSlot = 0.0f;
}

// ---------------------------------------------------------------------------
// Async staging: each wave copies 2 code rows (512B each) of the 16-code tile
// with GLOBAL_LOAD_ASYNC_TO_LDS_B128 in GVS mode (SGPR base + 32-bit VGPR
// offset; VDST = LDS byte address). Tracked by ASYNCcnt, no VGPR round-trip.
// ---------------------------------------------------------------------------
__device__ __forceinline__ void fill_tile_async(const _Float16* sBase,
                                                int buf,
                                                const _Float16* __restrict__ cbh,
                                                int tileCode0,   // first code id
                                                int wid, int lane) {
    uint32_t ldsBase = (uint32_t)(uintptr_t)sBase + (uint32_t)(buf * TILE_BYTES);
#pragma unroll
    for (int j = 0; j < 2; ++j) {
        const int code = wid * 2 + j;                               // 0..15
        uint32_t ldsOff = ldsBase + (uint32_t)(code * (BROW * 2) + lane * 16);
        uint32_t gOff   = (uint32_t)(((tileCode0 + code) << 9) + lane * 16); // 512B/row
        asm volatile("global_load_async_to_lds_b128 %0, %1, %2"
                     :: "v"(ldsOff), "v"(gOff), "s"(cbh)
                     : "memory");
    }
}

__device__ __forceinline__ void wait_async_all() {
    asm volatile("s_wait_asynccnt 0x0" ::: "memory");
}

// ---------------------------------------------------------------------------
// Kernel 3: partial argmin over a 1024-code segment via WMMA f16 (f32 accum).
// grid = (NTOK/256, KSPLIT); block = 256 threads = 8 waves, 32 tokens/wave
// (two 16-row A tile sets of (-2x) live in VGPRs for the whole loop).
// Per iteration: async-prefetch the next 16-code tile into the other LDS
// buffer, seed both accumulators with the broadcast ||e||^2 of this lane's
// code column, run 16 WMMAs (2 M-tiles x 8 K-chunks; each B fragment loaded
// once, used twice) so the final accumulator IS the distance surrogate
// ||e||^2 - 2 x.e, update running (min,idx), then one wait + one barrier.
// ---------------------------------------------------------------------------
__global__ __launch_bounds__(256, 1) void vq_argmin(
        const _Float16* __restrict__ xT,
        const _Float16* __restrict__ cbh,
        const float*    __restrict__ cnorm,
        float*          __restrict__ pmin,
        int*            __restrict__ pidx) {
    __shared__ _Float16 sB[2 * TILE_HALVES];        // double-buffered tiles

    const int tid  = threadIdx.x;
    const int lane = tid & 31;
    const int wid  = tid >> 5;
    const int mrow = lane & 15;                     // M for A, N for B/C/D
    const int hgrp = lane >> 4;                     // half-group select
    const int tok0 = blockIdx.x * BLK_TOK + wid * MWAVE;
    const int k0   = blockIdx.y * KSEG;             // this segment's first code

    // ---- persistent A fragments, 2 M-tiles (ISA 16-bit A 16x32 layout) ----
    ABfrag afrag[2][8];
    const int koff = hgrp * 8;
#pragma unroll
    for (int mt = 0; mt < 2; ++mt) {
        const _Float16* aBase = xT + (size_t)(tok0 + mt * 16 + mrow) * CC;
#pragma unroll
        for (int cc = 0; cc < 8; ++cc) {
            afrag[mt][cc].h[0] = *(const v8h*)(aBase + cc * 32 + koff);
            afrag[mt][cc].h[1] = *(const v8h*)(aBase + cc * 32 + 16 + koff);
        }
    }

    float runmin[2][8];
    int   runidx[2][8];
#pragma unroll
    for (int mt = 0; mt < 2; ++mt)
#pragma unroll
        for (int r = 0; r < 8; ++r) { runmin[mt][r] = 3.4e38f; runidx[mt][r] = 0; }

    const int kbB = hgrp * 8;                       // B K-offset per half-group

    // prologue: stage tile 0 into buffer 0
    fill_tile_async(sB, 0, cbh, k0, wid, lane);
    wait_async_all();
    __syncthreads();

    for (int kt = 0; kt < NITER; ++kt) {
        const int p = kt & 1;
        if (kt + 1 < NITER)                         // prefetch next tile
            fill_tile_async(sB, 1 - p, cbh, k0 + (kt + 1) * TCODES, wid, lane);

        const int   k  = k0 + kt * TCODES + mrow;   // this lane's code column
        const float cn = cnorm[k];

        // seed accumulators with ||e||^2 (same N column in all 8 slots)
        v8f acc0, acc1;
#pragma unroll
        for (int r = 0; r < 8; ++r) { acc0[r] = cn; acc1[r] = cn; }

        // ---- 8 K-chunks; each B fragment feeds both M-tiles ----
        const _Float16* bBase = sB + p * TILE_HALVES + mrow * BROW + kbB;
#pragma unroll
        for (int cc = 0; cc < 8; ++cc) {
            ABfrag bf;
            const _Float16* bp = bBase + cc * 32;
            bf.h[0] = *(const v8h*)(bp);
            bf.h[1] = *(const v8h*)(bp + 16);
            acc0 = __builtin_amdgcn_wmma_f32_16x16x32_f16(
                    false, afrag[0][cc].v, false, bf.v, (short)0, acc0, false, false);
            acc1 = __builtin_amdgcn_wmma_f32_16x16x32_f16(
                    false, afrag[1][cc].v, false, bf.v, (short)0, acc1, false, false);
        }

        // accumulators already hold ||e||^2 - 2 x.e
#pragma unroll
        for (int r = 0; r < 8; ++r) {
            if (acc0[r] < runmin[0][r]) { runmin[0][r] = acc0[r]; runidx[0][r] = k; }
            if (acc1[r] < runmin[1][r]) { runmin[1][r] = acc1[r]; runidx[1][r] = k; }
        }

        wait_async_all();                           // my prefetch landed
        __syncthreads();                            // everyone's landed + reads done
    }

    // ---- reduce across the 16 N-lanes (smaller index wins ties) ----
#pragma unroll
    for (int mt = 0; mt < 2; ++mt) {
#pragma unroll
        for (int r = 0; r < 8; ++r) {
            float v = runmin[mt][r];
            int   id = runidx[mt][r];
#pragma unroll
            for (int m = 1; m < 16; m <<= 1) {
                float ov = __shfl_xor(v, m, 32);
                int   oi = __shfl_xor(id, m, 32);
                if (ov < v || (ov == v && oi < id)) { v = ov; id = oi; }
            }
            runmin[mt][r] = v; runidx[mt][r] = id;
        }
    }

    if (mrow == 0) {
#pragma unroll
        for (int mt = 0; mt < 2; ++mt) {
#pragma unroll
            for (int r = 0; r < 8; ++r) {
                const int token = tok0 + mt * 16 + hgrp * 8 + r;
                pmin[blockIdx.y * NTOK + token] = runmin[mt][r];
                pidx[blockIdx.y * NTOK + token] = runidx[mt][r];
            }
        }
    }
}

// ---------------------------------------------------------------------------
// Kernel 4: merge the KSPLIT partial argmins (segments ascend in code id, so
// strict < plus index tie-break preserves first-occurrence semantics).
// ---------------------------------------------------------------------------
__global__ __launch_bounds__(256) void merge_idx(const float* __restrict__ pmin,
                                                 const int*   __restrict__ pidx,
                                                 int*   __restrict__ idxOut,
                                                 float* __restrict__ idxOutF) {
    const int n = blockIdx.x * 256 + threadIdx.x;
    float best = pmin[n];
    int   bi   = pidx[n];
#pragma unroll
    for (int s = 1; s < KSPLIT; ++s) {
        float v = pmin[s * NTOK + n];
        int  id = pidx[s * NTOK + n];
        if (v < best || (v == best && id < bi)) { best = v; bi = id; }
    }
    idxOut[n]  = bi;
    idxOutF[n] = (float)bi;
}

// ---------------------------------------------------------------------------
// Kernel 5: gather codebook rows back to NCHW (coalesced writes; codebook is
// L2-resident) + fused commitment-loss reduction (wave shuffle -> LDS -> one
// atomicAdd per block).
// ---------------------------------------------------------------------------
__global__ __launch_bounds__(256) void gather_loss(const float* __restrict__ x,
                                                   const float* __restrict__ cb,
                                                   const int*   __restrict__ idx,
                                                   float*       __restrict__ qe,
                                                   float*       __restrict__ lossSlot) {
    __shared__ float red[8];
    const int i = blockIdx.x * 256 + threadIdx.x;   // flat over B*C*H*W
    const int w = i & 31;
    const int h = (i >> 5) & 31;
    const int c = (i >> 10) & 255;
    const int b = i >> 18;
    const int n = (b << 10) + (h << 5) + w;

    const int k = idx[n];
    const float q = cb[(size_t)k * CC + c];
    qe[i] = q;                                      // straight-through fwd == q

    const float diff = x[i] - q;
    float p = diff * diff * (1.0f / ((float)NTOK * (float)CC));
#pragma unroll
    for (int m = 16; m >= 1; m >>= 1) p += __shfl_xor(p, m, 32);

    const int lane = threadIdx.x & 31;
    const int wid  = threadIdx.x >> 5;
    if (lane == 0) red[wid] = p;
    __syncthreads();
    if (threadIdx.x == 0) {
        float s = 0.0f;
#pragma unroll
        for (int j = 0; j < 8; ++j) s += red[j];
        atomicAdd(lossSlot, s);
    }
}

// ---------------------------------------------------------------------------
extern "C" void kernel_launch(void* const* d_in, const int* in_sizes, int n_in,
                              void* d_out, int out_size, void* d_ws, size_t ws_size,
                              hipStream_t stream) {
    (void)in_sizes; (void)n_in; (void)out_size; (void)ws_size;
    const float* x  = (const float*)d_in[0];        // [B,C,H,W] f32
    const float* cb = (const float*)d_in[1];        // [K,C] f32
    float* out = (float*)d_out;                     // qe | loss | indices(as f32)

    char* ws = (char*)d_ws;
    _Float16* xT    = (_Float16*)ws;                                            // 8 MB
    _Float16* cbh   = (_Float16*)(ws + ((size_t)8 << 20));                      // 4 MB
    float*    cnorm = (float*)   (ws + ((size_t)12 << 20));                     // 32 KB (64K slot)
    int*      idx   = (int*)     (ws + ((size_t)12 << 20) + (64 << 10));        // 64 KB
    float*    pmin  = (float*)   (ws + ((size_t)12 << 20) + (128 << 10));       // 512 KB
    int*      pidx  = (int*)     (ws + ((size_t)12 << 20) + (640 << 10));       // 512 KB

    prep_x   <<<QE_ELEMS / 256, 256, 0, stream>>>(x, xT);
    prep_cb  <<<KCODES / 8,     256, 0, stream>>>(cb, cbh, cnorm, out + LOSS_OFF);
    dim3 gmain(NTOK / BLK_TOK, KSPLIT);
    vq_argmin<<<gmain,          256, 0, stream>>>(xT, cbh, cnorm, pmin, pidx);
    merge_idx<<<NTOK / 256,     256, 0, stream>>>(pmin, pidx, idx, out + IDX_OFF);
    gather_loss<<<QE_ELEMS / 256, 256, 0, stream>>>(x, cb, idx, out, out + LOSS_OFF);
}
